// DualGNN_25546465477049
// MI455X (gfx1250) — matrix operand order repaired and checked by reading, not compiled, not run on hardware
//
#include <hip/hip_runtime.h>

typedef __attribute__((ext_vector_type(16))) _Float16 v16h;
typedef __attribute__((ext_vector_type(8)))  float    v8f;
typedef __attribute__((ext_vector_type(4)))  float    v4f;

#define NN      163840
#define HD      128
#define EE      1310720
#define BG      256
#define NEQV    128
#define NCL     16
#define NMU     768
#define NHTOT   (NN * HD)            // 20971520 floats

// ---------------------------------------------------------------------------
// Pack W_rel / W_root (L=3, each 128x128 f32 row-major, W[k][n]) into the
// V_WMMA_F32_16X16X32_F16 B-fragment layout:
//   frag(mat, nt, kt) : 32 lanes x 16 halfs contiguous
//   lane<16 : N = nt*16+lane,     K = kt*32 + {0..7, 16..23}[i]   (i=0..15)
//   lane>=16: N = nt*16+lane-16,  K = kt*32 + {8..15, 24..31}[i]
// flat = ((mat*8+nt)*4+kt)*512 + lane*16 + i
// ---------------------------------------------------------------------------
__global__ void dualgnn_pack_w(const float* __restrict__ Wrel,
                               const float* __restrict__ Wroot,
                               _Float16* __restrict__ Wp) {
    int f = blockIdx.x * blockDim.x + threadIdx.x;   // 98304 total
    int i    =  f        & 15;
    int lane = (f >> 4)  & 31;
    int kt   = (f >> 9)  & 3;
    int nt   = (f >> 11) & 7;
    int mat  =  f >> 14;            // 0..5
    int l    = mat >> 1;
    int sel  = mat & 1;
    const float* src = (sel ? Wroot : Wrel) + l * HD * HD;
    int k = kt * 32 + ((lane >= 16) ? 8 : 0) + (i & 7) + ((i & 8) << 1);
    int n = nt * 16 + (lane & 15);
    Wp[f] = (_Float16)src[k * HD + n];
}

// ---------------------------------------------------------------------------
// Encoder: h = relu(x @ W_enc + b_enc), x:[N,7], W_enc:[7,128]
// one wave per node, 4 columns per lane
// ---------------------------------------------------------------------------
__global__ void dualgnn_encoder(const float* __restrict__ x,
                                const float* __restrict__ Wenc,
                                const float* __restrict__ benc,
                                float* __restrict__ h) {
    int lane = threadIdx.x & 31;
    int wave = threadIdx.x >> 5;
    int node = blockIdx.x * 8 + wave;
    float xv[7];
#pragma unroll
    for (int k = 0; k < 7; ++k) xv[k] = x[node * 7 + k];
    int c0 = lane * 4;
    v4f acc;
#pragma unroll
    for (int j = 0; j < 4; ++j) acc[j] = benc[c0 + j];
#pragma unroll
    for (int k = 0; k < 7; ++k) {
        const float* wr = Wenc + k * HD + c0;
#pragma unroll
        for (int j = 0; j < 4; ++j) acc[j] += xv[k] * wr[j];
    }
#pragma unroll
    for (int j = 0; j < 4; ++j) acc[j] = fmaxf(acc[j], 0.f);
    *(v4f*)(h + (size_t)node * HD + c0) = acc;
}

// ---------------------------------------------------------------------------
// LayerNorm over H=128, one wave per node (float4 per lane + shfl_xor tree)
// ---------------------------------------------------------------------------
__global__ void dualgnn_layernorm(const float* __restrict__ h,
                                  const float* __restrict__ g,
                                  const float* __restrict__ bb,
                                  float* __restrict__ hln) {
    int lane = threadIdx.x & 31;
    int wave = threadIdx.x >> 5;
    int node = blockIdx.x * 8 + wave;
    int c0 = lane * 4;
    v4f v = *(const v4f*)(h + (size_t)node * HD + c0);
    float s  = v[0] + v[1] + v[2] + v[3];
    float sq = v[0]*v[0] + v[1]*v[1] + v[2]*v[2] + v[3]*v[3];
#pragma unroll
    for (int m = 16; m >= 1; m >>= 1) {
        s  += __shfl_xor(s,  m, 32);
        sq += __shfl_xor(sq, m, 32);
    }
    float mean = s * (1.0f / HD);
    float var  = sq * (1.0f / HD) - mean * mean;
    float rstd = rsqrtf(var + 1e-5f);
    v4f o;
#pragma unroll
    for (int j = 0; j < 4; ++j)
        o[j] = (v[j] - mean) * rstd * g[c0 + j] + bb[c0 + j];
    *(v4f*)(hln + (size_t)node * HD + c0) = o;
}

__global__ void dualgnn_zero(float* __restrict__ p) {
    int i = blockIdx.x * blockDim.x + threadIdx.x;   // NHTOT/4 threads
    v4f z{0.f, 0.f, 0.f, 0.f};
    *(v4f*)(p + (size_t)i * 4) = z;
}

// ---------------------------------------------------------------------------
// Edge scatter: agg[dst] += hln[src]; one wave per edge, 4 f32 atomics/lane.
// hln/agg are L2-resident (84MB each, 192MB L2).
// ---------------------------------------------------------------------------
__global__ void dualgnn_scatter(const long long* __restrict__ ei,
                                const float* __restrict__ hln,
                                float* __restrict__ agg) {
    int lane = threadIdx.x & 31;
    int wave = threadIdx.x >> 5;
    long long e = (long long)blockIdx.x * 8 + wave;
    long long s = ei[e];
    long long d = ei[(long long)EE + e];
    int c0 = lane * 4;
    v4f v = *(const v4f*)(hln + s * HD + c0);
    float* ap = agg + d * HD + c0;
    atomicAdd(ap + 0, v[0]);
    atomicAdd(ap + 1, v[1]);
    atomicAdd(ap + 2, v[2]);
    atomicAdd(ap + 3, v[3]);
}

// ---------------------------------------------------------------------------
// A-fragment load for V_WMMA_F32_16X16X32_F16 from row-major f32 [*,128]:
//   lane<16 : M = rowbase+lane,    halfs = K {k0..k0+7, k0+16..k0+23}
//   lane>=16: M = rowbase+lane-16, halfs = K {k0+8..+15, k0+24..+31}
// ---------------------------------------------------------------------------
__device__ inline v16h dualgnn_load_a(const float* __restrict__ M,
                                      int rowbase, int k0, int lane) {
    int row  = rowbase + (lane & 15);
    int koff = k0 + ((lane >> 4) << 3);
    const float* p = M + (size_t)row * HD + koff;
    v4f f0 = *(const v4f*)(p);
    v4f f1 = *(const v4f*)(p + 4);
    v4f f2 = *(const v4f*)(p + 16);
    v4f f3 = *(const v4f*)(p + 20);
    v16h a;
#pragma unroll
    for (int i = 0; i < 4; ++i) {
        a[i]      = (_Float16)f0[i];
        a[4 + i]  = (_Float16)f1[i];
        a[8 + i]  = (_Float16)f2[i];
        a[12 + i] = (_Float16)f3[i];
    }
    return a;
}

// ---------------------------------------------------------------------------
// Fused GraphConv GEMM: h_out = relu(agg @ W_rel[l] + hln @ W_root[l] + b_rel[l])
// 8 waves/block, one 16-row tile per wave; A fragments (agg + hln, K=128) kept
// in VGPRs; 8 column tiles x 4 K-steps x 2 matrices = 64 WMMAs per wave.
// ---------------------------------------------------------------------------
__global__ void dualgnn_gemm(const float* __restrict__ agg,
                             const float* __restrict__ hln,
                             const _Float16* __restrict__ Wp,
                             const float* __restrict__ brel,
                             float* __restrict__ hout,
                             int layer) {
    int lane = threadIdx.x & 31;
    int wave = threadIdx.x >> 5;
    int rowbase = (blockIdx.x * 8 + wave) * 16;

    const _Float16* Wrel_l  = Wp + (size_t)(layer * 2 + 0) * 16384;
    const _Float16* Wroot_l = Wp + (size_t)(layer * 2 + 1) * 16384;
    const float* bias_l = brel + layer * HD;

    v16h aA[4], aR[4];
#pragma unroll
    for (int kt = 0; kt < 4; ++kt) {
        aA[kt] = dualgnn_load_a(agg, rowbase, kt * 32, lane);
        aR[kt] = dualgnn_load_a(hln, rowbase, kt * 32, lane);
    }

    int nlo   = lane & 15;
    int hiRow = (lane >> 4) * 8;
#pragma unroll
    for (int nt = 0; nt < 8; ++nt) {
        v8f c{0.f, 0.f, 0.f, 0.f, 0.f, 0.f, 0.f, 0.f};
#pragma unroll
        for (int kt = 0; kt < 4; ++kt) {
            v16h b0 = *(const v16h*)(Wrel_l  + (nt * 4 + kt) * 512 + lane * 16);
            c = __builtin_amdgcn_wmma_f32_16x16x32_f16(false, aA[kt], false, b0,
                                                       (short)0, c, false, false);
            v16h b1 = *(const v16h*)(Wroot_l + (nt * 4 + kt) * 512 + lane * 16);
            c = __builtin_amdgcn_wmma_f32_16x16x32_f16(false, aR[kt], false, b1,
                                                       (short)0, c, false, false);
        }
        int n = nt * 16 + nlo;
        float bv = bias_l[n];
#pragma unroll
        for (int j = 0; j < 8; ++j) {
            int row = rowbase + hiRow + j;
            hout[(size_t)row * HD + n] = fmaxf(c[j] + bv, 0.f);
        }
    }
}

// ---------------------------------------------------------------------------
// Head: logits -> softmax -> lamb, one thread per loc row (32768 rows).
// W_head (128x16) staged in LDS.
// ---------------------------------------------------------------------------
__global__ void dualgnn_head(const float* __restrict__ h,
                             const long long* __restrict__ loc,
                             const float* __restrict__ Whead,
                             const float* __restrict__ bhead,
                             const float* __restrict__ classes,
                             float* __restrict__ lamb) {
    __shared__ float sW[HD * NCL];
    __shared__ float sB[NCL];
    __shared__ float sC[NCL];
    int tid = threadIdx.x;
#pragma unroll
    for (int t = 0; t < 8; ++t) sW[tid * 8 + t] = Whead[tid * 8 + t];
    if (tid < NCL) { sB[tid] = bhead[tid]; sC[tid] = classes[tid]; }
    __syncthreads();

    int row = blockIdx.x * blockDim.x + tid;         // 0..32767
    long long node = loc[row];
    const float* hp = h + (size_t)node * HD;

    float acc[NCL];
#pragma unroll
    for (int c = 0; c < NCL; ++c) acc[c] = sB[c];
    for (int k = 0; k < HD; ++k) {
        float hv = hp[k];
        const float* wr = &sW[k * NCL];
#pragma unroll
        for (int c = 0; c < NCL; ++c) acc[c] += hv * wr[c];
    }
    float m = acc[0];
#pragma unroll
    for (int c = 1; c < NCL; ++c) m = fmaxf(m, acc[c]);
    float sum = 0.f, lam = 0.f;
#pragma unroll
    for (int c = 0; c < NCL; ++c) {
        float e = __expf(acc[c] - m);
        sum += e;
        lam += e * sC[c];
    }
    lamb[row] = lam / sum;
}

// ---------------------------------------------------------------------------
// KKT: mu = obj + lamb @ eq_cm; out_mu = interleaved relu(mu)/relu(-mu) blocks
// one thread per (graph, j) pair.
// ---------------------------------------------------------------------------
__global__ void dualgnn_kkt(const float* __restrict__ lamb,
                            const float* __restrict__ eqcm,
                            const float* __restrict__ obj,
                            float* __restrict__ outmu) {
    int t  = blockIdx.x * blockDim.x + threadIdx.x;  // 0..196607
    int bi = t / NMU;
    int j  = t % NMU;
    float acc = obj[j];
    const float* lp = lamb + bi * NEQV;
    for (int k = 0; k < NEQV; ++k)
        acc += lp[k] * eqcm[k * NMU + j];
    float lb = fmaxf(acc, 0.f);
    float ub = fmaxf(-acc, 0.f);
    int seg = j >> 8;                                // 0,1,2
    float* ob = outmu + (size_t)bi * (2 * NMU);
    ob[j + 256 * seg]       = lb;
    ob[j + 256 * (seg + 1)] = ub;
}

// ---------------------------------------------------------------------------
extern "C" void kernel_launch(void* const* d_in, const int* in_sizes, int n_in,
                              void* d_out, int out_size, void* d_ws, size_t ws_size,
                              hipStream_t stream) {
    const float*     x       = (const float*)d_in[0];
    const long long* ei      = (const long long*)d_in[1];
    const long long* loc     = (const long long*)d_in[2];
    const float*     Wenc    = (const float*)d_in[3];
    const float*     benc    = (const float*)d_in[4];
    const float*     Wrel    = (const float*)d_in[5];
    const float*     brel    = (const float*)d_in[6];
    const float*     Wroot   = (const float*)d_in[7];
    const float*     lng     = (const float*)d_in[8];
    const float*     lnb     = (const float*)d_in[9];
    const float*     Whead   = (const float*)d_in[10];
    const float*     bhead   = (const float*)d_in[11];
    const float*     classes = (const float*)d_in[12];
    const float*     eqcm    = (const float*)d_in[13];
    const float*     obj     = (const float*)d_in[14];

    float* out   = (float*)d_out;
    float* outmu = out;                      // [256, 1536]
    float* lamb  = out + BG * 2 * NMU;       // [256, 128]

    float* ws   = (float*)d_ws;
    float* h    = ws;
    float* hln  = h + NHTOT;
    float* agg  = hln + NHTOT;
    _Float16* Wp = (_Float16*)(agg + NHTOT);

    dualgnn_pack_w <<<384,   256, 0, stream>>>(Wrel, Wroot, Wp);
    dualgnn_encoder<<<20480, 256, 0, stream>>>(x, Wenc, benc, h);

    for (int l = 0; l < 3; ++l) {
        dualgnn_layernorm<<<20480,  256, 0, stream>>>(h, lng, lnb, hln);
        dualgnn_zero     <<<20480,  256, 0, stream>>>(agg);
        dualgnn_scatter  <<<163840, 256, 0, stream>>>(ei, hln, agg);
        dualgnn_gemm     <<<1280,   256, 0, stream>>>(agg, hln, Wp, brel, h, l);
    }

    dualgnn_head<<<128, 256, 0, stream>>>(h, loc, Whead, bhead, classes, lamb);
    dualgnn_kkt <<<768, 256, 0, stream>>>(lamb, eqcm, obj, outmu);
}
